// GraphSAGE_81664508166317
// MI455X (gfx1250) — compile-verified, matrix operand order checked by reading, not run on hardware
//
#include <hip/hip_runtime.h>
#include <math.h>

// ---------------- problem constants (match reference) ----------------
#define N_NODES  50000
#define N_EDGES  600000
#define IN_SIZE  128
#define HIDDEN   256
#define OUT_SIZE 128

typedef __attribute__((ext_vector_type(16))) __bf16 v16bf;
typedef __attribute__((ext_vector_type(8)))  __bf16 v8bf;
typedef __attribute__((ext_vector_type(8)))  float  v8f;

// ---------------- utility kernels ----------------
__global__ __launch_bounds__(256)
void zero_f32(float* __restrict__ p, int n) {
    int i = blockIdx.x * blockDim.x + threadIdx.x;
    if (i < n) p[i] = 0.0f;
}

// Wcat[r][0:kHalf] = Wl[r], Wcat[r][kHalf:2*kHalf] = Wr[r]   (bf16 pack)
__global__ __launch_bounds__(256)
void build_wcat(const float* __restrict__ Wl, const float* __restrict__ Wr,
                __bf16* __restrict__ Wcat, int rows, int kHalf) {
    int i = blockIdx.x * blockDim.x + threadIdx.x;
    if (i >= rows * kHalf) return;
    int r = i / kHalf, c = i % kHalf;
    __bf16* row = Wcat + (size_t)r * (2 * kHalf);
    row[c]         = (__bf16)Wl[i];
    row[kHalf + c] = (__bf16)Wr[i];
}

// ---------------- edge scatter (mean-aggregate numerator) ----------------
// one wave per edge; coalesced float4 gather of X[src], f32 atomics into agg[dst].
// agg buffers (25.6 / 51.2 MB) fit in the 192MB L2, so the RMWs stay on-chip.
__global__ __launch_bounds__(256)
void scatter_add(const float* __restrict__ X, const int* __restrict__ src,
                 const int* __restrict__ dst, float* __restrict__ agg,
                 float* __restrict__ cnt, int nedges, int F) {
    int wave = blockIdx.x * (blockDim.x >> 5) + (threadIdx.x >> 5);
    int lane = threadIdx.x & 31;
    if (wave >= nedges) return;
    int s = src[wave];
    int d = dst[wave];
    if (cnt != nullptr && lane == 0) atomicAdd(cnt + d, 1.0f);
    const float4* xs = (const float4*)(X + (size_t)s * F);
    float* ad = agg + (size_t)d * F;
    int nvec = F >> 2;
    for (int i = lane; i < nvec; i += 32) {
        float4 v = xs[i];
        atomicAdd(ad + i * 4 + 0, v.x);
        atomicAdd(ad + i * 4 + 1, v.y);
        atomicAdd(ad + i * 4 + 2, v.z);
        atomicAdd(ad + i * 4 + 3, v.w);
    }
}

// Xcat1[n][0:F] = bf16(agg[n]/max(cnt,1)),  Xcat1[n][F:2F] = bf16(feat[n])
__global__ __launch_bounds__(256)
void prep_xcat1(const float* __restrict__ agg, const float* __restrict__ cnt,
                const float* __restrict__ feat, __bf16* __restrict__ X,
                int nnodes, int F) {
    int i = blockIdx.x * blockDim.x + threadIdx.x;
    if (i >= nnodes * F) return;
    int n = i / F, c = i % F;
    float inv = 1.0f / fmaxf(cnt[n], 1.0f);
    __bf16* row = X + (size_t)n * (2 * F);
    row[c]     = (__bf16)(agg[i] * inv);
    row[F + c] = (__bf16)feat[i];
}

// Xcat2[n][0:F] = bf16(agg2[n]/max(cnt,1))   (right half written by norm_relu_store)
__global__ __launch_bounds__(256)
void prep_xcat2(const float* __restrict__ agg, const float* __restrict__ cnt,
                __bf16* __restrict__ X, int nnodes, int F) {
    int i = blockIdx.x * blockDim.x + threadIdx.x;
    if (i >= nnodes * F) return;
    int n = i / F, c = i % F;
    float inv = 1.0f / fmaxf(cnt[n], 1.0f);
    X[(size_t)n * (2 * F) + c] = (__bf16)(agg[i] * inv);
}

// ---------------- bf16 WMMA GEMM:  C[M,N] = A[M,K] * B[N,K]^T + bias ----------------
// One wave computes a 16 x (16*NT) output strip: the A fragment is loaded once per
// k-step and reused by NT WMMAs feeding NT independent accumulators (pipelines the
// matrix unit past the bf16 WMMA->WMMA RAW hazard and cuts A traffic NT-fold).
// Operand layout per CDNA5 ISA 7.12.2 (A 16x32 bf16, B assumed symmetric):
//   lane l: row/col = (l&15); k-half = (l>>4)*8; elements {kh..kh+7, kh+16..kh+23}
// All operand loads are 16B (global_load_b128); K templated so the loop fully unrolls.
template <int K, int NT>
__global__ __launch_bounds__(256)
void gemm_bf16_wmma(const __bf16* __restrict__ A, const __bf16* __restrict__ B,
                    const float* __restrict__ bias, float* __restrict__ C,
                    int M, int N) {
    const int lane    = threadIdx.x & 31;
    const int wave    = blockIdx.x * (blockDim.x >> 5) + (threadIdx.x >> 5);
    const int ngroups = N / (16 * NT);
    if (wave >= (M >> 4) * ngroups) return;
    const int tm  = (wave / ngroups) << 4;
    const int tn0 = (wave % ngroups) * (16 * NT);
    const int kh  = (lane >> 4) * 8;   // k-half element offset: 0 or 8
    const int l15 = lane & 15;

    const __bf16* arow = A + (size_t)(tm  + l15) * K + kh;
    const __bf16* brow = B + (size_t)(tn0 + l15) * K + kh;  // tiles at +t*16*K

    v8f acc[NT];
#pragma unroll
    for (int t = 0; t < NT; ++t) acc[t] = (v8f){};

#pragma unroll
    for (int k0 = 0; k0 < K; k0 += 32) {
        union { v16bf v; v8bf h[2]; } a;
        a.h[0] = *(const v8bf*)(arow + k0);
        a.h[1] = *(const v8bf*)(arow + k0 + 16);
#pragma unroll
        for (int t = 0; t < NT; ++t) {
            union { v16bf v; v8bf h[2]; } b;
            b.h[0] = *(const v8bf*)(brow + (size_t)t * 16 * K + k0);
            b.h[1] = *(const v8bf*)(brow + (size_t)t * 16 * K + k0 + 16);
            acc[t] = __builtin_amdgcn_wmma_f32_16x16x32_bf16(
                false, a.v, false, b.v, (short)0, acc[t], false, false);
        }
    }

    const int r0 = tm + (lane >> 4) * 8;   // C/D layout: VGPR v -> row r0+v, col l15
#pragma unroll
    for (int t = 0; t < NT; ++t) {
        const int cn = tn0 + t * 16 + l15;
        const float bc = bias[cn];
#pragma unroll
        for (int v = 0; v < 8; ++v)
            C[(size_t)(r0 + v) * N + cn] = acc[t][v] + bc;
    }
}

// ---------------- row L2-normalize + ReLU; also emit bf16 into Xcat2 right half ----
// one block (256 threads) per node; F must equal blockDim.x
__global__ __launch_bounds__(256)
void norm_relu_store(float* __restrict__ H, __bf16* __restrict__ X2,
                     int F, int ldx, int coloff) {
    __shared__ float red[256];
    const int n = blockIdx.x;
    const int t = threadIdx.x;
    float* row = H + (size_t)n * F;
    float v = row[t];
    red[t] = v * v;
    __syncthreads();
    for (int s = 128; s > 0; s >>= 1) {
        if (t < s) red[t] += red[t + s];
        __syncthreads();
    }
    float inv = 1.0f / fmaxf(sqrtf(red[0]), 1e-12f);
    float r = fmaxf(v * inv, 0.0f);          // normalize then ReLU (reference order)
    row[t] = r;
    X2[(size_t)n * ldx + coloff + t] = (__bf16)r;
}

// ---------------- final: normalize h2, logits = h2n @ Wfc^T + bfc, softmax --------
// one wave per node; wave32 shuffle reduction
__global__ __launch_bounds__(256)
void final_fc(const float* __restrict__ H2, const float* __restrict__ Wfc,
              const float* __restrict__ bfc, float* __restrict__ out,
              int nnodes, int F) {
    int wave = blockIdx.x * (blockDim.x >> 5) + (threadIdx.x >> 5);
    int lane = threadIdx.x & 31;
    if (wave >= nnodes) return;
    const float* h = H2 + (size_t)wave * F;
    float ss = 0.0f, d0 = 0.0f, d1 = 0.0f;
    for (int i = lane; i < F; i += 32) {
        float v = h[i];
        ss += v * v;
        d0 += v * Wfc[i];
        d1 += v * Wfc[F + i];
    }
    for (int off = 16; off > 0; off >>= 1) {
        ss += __shfl_xor(ss, off, 32);
        d0 += __shfl_xor(d0, off, 32);
        d1 += __shfl_xor(d1, off, 32);
    }
    if (lane == 0) {
        float inv = 1.0f / fmaxf(sqrtf(ss), 1e-12f);
        float l0 = d0 * inv + bfc[0];
        float l1 = d1 * inv + bfc[1];
        float m  = fmaxf(l0, l1);
        float e0 = __expf(l0 - m), e1 = __expf(l1 - m);
        float s  = 1.0f / (e0 + e1);
        out[(size_t)wave * 2 + 0] = e0 * s;
        out[(size_t)wave * 2 + 1] = e1 * s;
    }
}

// ---------------- host-side orchestration ----------------
extern "C" void kernel_launch(void* const* d_in, const int* in_sizes, int n_in,
                              void* d_out, int out_size, void* d_ws, size_t ws_size,
                              hipStream_t stream) {
    (void)in_sizes; (void)n_in; (void)out_size; (void)ws_size;

    const float* feat = (const float*)d_in[0];
    const int*   eidx = (const int*)  d_in[1];
    const int*   src  = eidx;                 // edge_index[0]
    const int*   dst  = eidx + N_EDGES;       // edge_index[1]
    const float* W1l  = (const float*)d_in[2];
    const float* b1   = (const float*)d_in[3];
    const float* W1r  = (const float*)d_in[4];
    const float* W2l  = (const float*)d_in[5];
    const float* b2   = (const float*)d_in[6];
    const float* W2r  = (const float*)d_in[7];
    const float* Wfc  = (const float*)d_in[8];
    const float* bfc  = (const float*)d_in[9];
    float* out = (float*)d_out;

    // bump allocator over workspace (256B aligned sub-buffers; ~231 MB total)
    char* ws = (char*)d_ws;
    size_t off = 0;
    auto alloc = [&](size_t bytes) -> void* {
        void* p = ws + off;
        off = (off + bytes + 255) & ~(size_t)255;
        return p;
    };
    float*  cnt   = (float*) alloc((size_t)N_NODES * 4);                 // in-degree
    float*  agg1  = (float*) alloc((size_t)N_NODES * IN_SIZE * 4);       // layer1 sum
    __bf16* Xcat1 = (__bf16*)alloc((size_t)N_NODES * 2 * IN_SIZE * 2);   // [agg_mean|x]
    __bf16* Wcat1 = (__bf16*)alloc((size_t)HIDDEN * 2 * IN_SIZE * 2);    // [W1l;W1r]
    float*  h1    = (float*) alloc((size_t)N_NODES * HIDDEN * 4);        // pre/post norm
    float*  agg2  = (float*) alloc((size_t)N_NODES * HIDDEN * 4);        // layer2 sum
    __bf16* Xcat2 = (__bf16*)alloc((size_t)N_NODES * 2 * HIDDEN * 2);    // [agg2_mean|h]
    __bf16* Wcat2 = (__bf16*)alloc((size_t)OUT_SIZE * 2 * HIDDEN * 2);   // [W2l;W2r]
    float*  h2    = (float*) alloc((size_t)N_NODES * OUT_SIZE * 4);

    const int TB = 256;
    auto cdiv = [](int a, int b) { return (a + b - 1) / b; };

    // 1) zero the atomic accumulators (workspace is poisoned, and must be re-zeroed
    //    every call for determinism)
    zero_f32<<<cdiv(N_NODES, TB), TB, 0, stream>>>(cnt, N_NODES);
    zero_f32<<<cdiv(N_NODES * IN_SIZE, TB), TB, 0, stream>>>(agg1, N_NODES * IN_SIZE);
    zero_f32<<<cdiv(N_NODES * HIDDEN, TB), TB, 0, stream>>>(agg2, N_NODES * HIDDEN);

    // 2) bf16 weight concats
    build_wcat<<<cdiv(HIDDEN * IN_SIZE, TB), TB, 0, stream>>>(W1l, W1r, Wcat1, HIDDEN, IN_SIZE);
    build_wcat<<<cdiv(OUT_SIZE * HIDDEN, TB), TB, 0, stream>>>(W2l, W2r, Wcat2, OUT_SIZE, HIDDEN);

    // 3) layer-1 scatter (numerator + counts); one wave per edge
    scatter_add<<<cdiv(N_EDGES, TB / 32), TB, 0, stream>>>(feat, src, dst, agg1, cnt,
                                                           N_EDGES, IN_SIZE);
    // 4) build bf16 Xcat1 = [agg1/cnt | feat]
    prep_xcat1<<<cdiv(N_NODES * IN_SIZE, TB), TB, 0, stream>>>(agg1, cnt, feat, Xcat1,
                                                               N_NODES, IN_SIZE);
    // 5) layer-1 fused GEMM: h1 = Xcat1 @ Wcat1^T + b1   (M=50000, N=256, K=256)
    {
        int waves = (N_NODES / 16) * (HIDDEN / 64);   // 16x64 strip per wave
        gemm_bf16_wmma<2 * IN_SIZE, 4><<<cdiv(waves, TB / 32), TB, 0, stream>>>(
            Xcat1, Wcat1, b1, h1, N_NODES, HIDDEN);
    }
    // 6) row L2-normalize + ReLU; also write bf16 h into Xcat2 right half
    norm_relu_store<<<N_NODES, HIDDEN, 0, stream>>>(h1, Xcat2, HIDDEN, 2 * HIDDEN, HIDDEN);

    // 7) layer-2 scatter (counts unchanged -> reuse cnt)
    scatter_add<<<cdiv(N_EDGES, TB / 32), TB, 0, stream>>>(h1, src, dst, agg2, nullptr,
                                                           N_EDGES, HIDDEN);
    // 8) Xcat2 left half = agg2/cnt
    prep_xcat2<<<cdiv(N_NODES * HIDDEN, TB), TB, 0, stream>>>(agg2, cnt, Xcat2,
                                                              N_NODES, HIDDEN);
    // 9) layer-2 fused GEMM: h2 = Xcat2 @ Wcat2^T + b2   (M=50000, N=128, K=512)
    {
        int waves = (N_NODES / 16) * (OUT_SIZE / 64);  // 16x64 strip per wave
        gemm_bf16_wmma<2 * HIDDEN, 4><<<cdiv(waves, TB / 32), TB, 0, stream>>>(
            Xcat2, Wcat2, b2, h2, N_NODES, OUT_SIZE);
    }
    // 10) normalize h2, tiny FC, softmax -> out
    final_fc<<<cdiv(N_NODES, TB / 32), TB, 0, stream>>>(h2, Wfc, bfc, out,
                                                        N_NODES, OUT_SIZE);
}
